// AbstractBlast_76733885710857
// MI455X (gfx1250) — matrix operand order repaired and checked by reading, not compiled, not run on hardware
//
#include <hip/hip_runtime.h>
#include <hip/hip_bf16.h>

// ---------------------------------------------------------------------------
// BLAST block-low-rank linear:  Y = bias + X @ W,
//   W_block(j,o) = (Vt[j] * S[o,j,:]) @ U[o]
// Factored pipeline (18 GFLOP, ~300 MB HBM traffic -> bandwidth bound):
//   k1: T[j, row, r]   = sum_k X[row, j*256+k] * Vt[j][k][r]        (WMMA bf16)
//   k2: Z[row, r]      = sum_j T[j, row, r] * S[o, j, r]            (VALU, with
//                        async double-buffered LDS staging of T slabs)
//       Y[row, o*256+p]= bias[o*256+p] + sum_r Z[row,r]*U[o][r][p]  (WMMA bf16)
// T kept in bf16, j-major, in workspace (32 MB -> L2 resident on 192 MB L2).
// X loads / Y stores are non-temporal so the streaming 268 MB does not evict T.
// ---------------------------------------------------------------------------

typedef __bf16 bf16_t;
typedef __attribute__((ext_vector_type(16))) __bf16 v16bf;
typedef __attribute__((ext_vector_type(8)))  float  v8f;

static constexpr int IN_DIM  = 4096;
static constexpr int OUT_DIM = 4096;
static constexpr int M_ROWS  = 8192;  // B*T
static constexpr int NJ      = 16;    // input blocks
static constexpr int BS      = 256;   // block size (in and out)
static constexpr int RK      = 128;   // rank

// ---- CDNA5 async global->LDS path (guarded; falls back to plain copy) -----
#if defined(__has_builtin)
#if __has_builtin(__builtin_amdgcn_global_load_async_to_lds_b128)
#define HAS_ASYNC_LDS 1
#endif
#endif
#ifndef HAS_ASYNC_LDS
#define HAS_ASYNC_LDS 0
#endif

#if HAS_ASYNC_LDS
#if __has_builtin(__builtin_amdgcn_s_wait_asynccnt)
#define WAIT_ASYNC(n) __builtin_amdgcn_s_wait_asynccnt(n)
#else
#define WAIT_ASYNC(n) asm volatile("s_wait_asynccnt %0" ::"i"(n) : "memory")
#endif
#else
#define WAIT_ASYNC(n) ((void)0)
#endif

// 16-byte vector type matching the async-to-LDS builtin's pointer params
typedef int v4i_b128 __attribute__((vector_size(16)));
typedef __attribute__((address_space(1))) v4i_b128* gb128_t;
typedef __attribute__((address_space(3))) v4i_b128* lb128_t;

// Copy (or async-issue the copy of) one contiguous 32 KB T slab into LDS.
// 16384 bf16 = 2048 x 16B chunks; 256 threads x 8 chunks, fully coalesced.
__device__ __forceinline__ void slab_copy_issue(const bf16_t* __restrict__ g,
                                                bf16_t* l, int tid) {
#if HAS_ASYNC_LDS
    gb128_t gp = (gb128_t)g;
    lb128_t lp = (lb128_t)l;
#pragma unroll
    for (int c = 0; c < 8; ++c) {
        const int i = c * 256 + tid;
        __builtin_amdgcn_global_load_async_to_lds_b128(gp + i, lp + i, 0, 0);
    }
#else
    const float4* g4 = (const float4*)g;
    float4* l4 = (float4*)l;
#pragma unroll
    for (int c = 0; c < 8; ++c) {
        const int i = c * 256 + tid;
        l4[i] = g4[i];
    }
#endif
}

// ---------------------------------------------------------------------------
// Kernel 1: T_tile(128 x 128) = X(128 x 256, block j) @ Vt[j](256 x 128)
// grid = (64 row-tiles, 16 j), block = 256 threads (8 waves)
// ---------------------------------------------------------------------------
__global__ __launch_bounds__(256) void blast_k1(const float* __restrict__ X,
                                                const float* __restrict__ Vt,
                                                bf16_t* __restrict__ T) {
    const int jt   = blockIdx.y;
    const int row0 = blockIdx.x * 128;

    // Vt[j] transposed into LDS as bf16: VtT[r][k]
    __shared__ bf16_t VtT[RK * BS];   // 64 KB

    const int tid = threadIdx.x;
    const float* Vj = Vt + (size_t)jt * BS * RK;     // row-major [k][r]
    for (int idx = tid; idx < BS * RK; idx += 256) {
        const int k = idx >> 7;
        const int r = idx & 127;
        VtT[r * BS + k] = (bf16_t)Vj[idx];
    }
    __syncthreads();

    const int wave  = tid >> 5;
    const int lane  = tid & 31;
    const int l16   = lane & 15;
    const int lhalf = lane >> 4;

    const float* xrow = X + (size_t)(row0 + wave * 16 + l16) * IN_DIM + jt * BS;

    v8f acc[8];
#pragma unroll
    for (int nt = 0; nt < 8; ++nt)
#pragma unroll
        for (int e = 0; e < 8; ++e) acc[nt][e] = 0.0f;

    for (int k0 = 0; k0 < BS; k0 += 32) {
        // A fragment (contiguous k = 16*lhalf + i, matching B); X is streamed -> NT
        v16bf a;
        const float* p0 = xrow + k0 + lhalf * 16;
        const float* p1 = p0 + 8;
#pragma unroll
        for (int i = 0; i < 8; ++i) {
            a[i]     = (bf16_t)__builtin_nontemporal_load(p0 + i);
            a[8 + i] = (bf16_t)__builtin_nontemporal_load(p1 + i);
        }
#pragma unroll
        for (int nt = 0; nt < 8; ++nt) {
            v16bf b;
            const bf16_t* bp = &VtT[(nt * 16 + l16) * BS + k0 + lhalf * 16];
#pragma unroll
            for (int i = 0; i < 16; ++i) b[i] = bp[i];
            acc[nt] = __builtin_amdgcn_wmma_f32_16x16x32_bf16(
                false, a, false, b, (short)0, acc[nt], false, false);
        }
    }

    // Store T tile as bf16, j-major layout T[j][row][r]  (L2-resident, RT)
    bf16_t* trow =
        T + ((size_t)jt * M_ROWS + row0 + wave * 16 + lhalf * 8) * RK;
#pragma unroll
    for (int nt = 0; nt < 8; ++nt) {
        const int c = nt * 16 + l16;
#pragma unroll
        for (int m = 0; m < 8; ++m)
            trow[(size_t)m * RK + c] = (bf16_t)acc[nt][m];
    }
}

// ---------------------------------------------------------------------------
// Kernel 2: per (row-tile, o):
//   Z(128 x 128) = sum_j T[j]_slab * S[o, j, :]   (async LDS staging + VALU)
//   Y_tile(128 x 256) = Z @ U[o] + bias           (WMMA bf16)
// grid = (64 row-tiles, 16 o), block = 256 threads (8 waves)
// ---------------------------------------------------------------------------
__global__ __launch_bounds__(256) void blast_k2(const bf16_t* __restrict__ T,
                                                const float* __restrict__ S,
                                                const float* __restrict__ U,
                                                const float* __restrict__ bias,
                                                float* __restrict__ Y) {
    const int o    = blockIdx.y;
    const int row0 = blockIdx.x * 128;
    const int tid  = threadIdx.x;

    // Shared arena: phase A uses two 32 KB T slabs at [0,64K);
    // phase B uses UT (64 KB) at [32K,96K) (slab reads are done by then).
    __shared__ __align__(16) char arena[96 * 1024];
    __shared__ float  Sl[NJ * RK];    //  8 KB
    __shared__ bf16_t Zb[128 * RK];   // 32 KB
    // NOTE: slab pointers are computed inline (arena + buf*32768); a local
    // pointer array would create a static initializer with an LDS
    // addrspacecast, which ld.lld cannot relocate.
    bf16_t* UT = (bf16_t*)(arena + 32768);   // [p][r]

    const float* So = S + (size_t)o * NJ * RK;
    for (int idx = tid; idx < NJ * RK; idx += 256) Sl[idx] = So[idx];

    // ---- Phase A: Z accumulation with double-buffered slab staging ----
    const bf16_t* Tbase = T + (size_t)row0 * RK;   // + j*M_ROWS*RK per slab
    slab_copy_issue(Tbase, (bf16_t*)(arena + 0), tid);

    const int rowB = tid >> 4;          // 0..15, +16*g covers 128 rows
    const int seg  = (tid & 15) * 8;    // 8 contiguous r per group

    float zacc[8][8];
#pragma unroll
    for (int g = 0; g < 8; ++g)
#pragma unroll
        for (int i = 0; i < 8; ++i) zacc[g][i] = 0.0f;

    for (int j = 0; j < NJ; ++j) {
        const int buf = j & 1;
        if (j + 1 < NJ) {
            slab_copy_issue(Tbase + (size_t)(j + 1) * M_ROWS * RK,
                            (bf16_t*)(arena + (size_t)(buf ^ 1) * 32768), tid);
            WAIT_ASYNC(8);   // slab j done; slab j+1 may stay in flight
        } else {
            WAIT_ASYNC(0);
        }
        __syncthreads();

        const bf16_t* sl = (const bf16_t*)(arena + (size_t)buf * 32768);
#pragma unroll
        for (int g = 0; g < 8; ++g) {
            const bf16_t* p  = sl + (rowB + 16 * g) * RK + seg;
            const float*  sp = &Sl[j * RK + seg];
#pragma unroll
            for (int i = 0; i < 8; ++i)
                zacc[g][i] += (float)p[i] * sp[i];
        }
        __syncthreads();   // all reads of this buffer done before reuse
    }

    // ---- Stage U[o]^T into LDS (overlaps retired slab space) + write Z ----
    const float* Uo = U + (size_t)o * RK * BS;       // row-major [r][p]
    for (int idx = tid; idx < RK * BS; idx += 256) {
        const int r = idx >> 8;
        const int p = idx & 255;
        UT[p * RK + r] = (bf16_t)Uo[idx];
    }
#pragma unroll
    for (int g = 0; g < 8; ++g)
#pragma unroll
        for (int i = 0; i < 8; ++i)
            Zb[(rowB + 16 * g) * RK + seg + i] = (bf16_t)zacc[g][i];
    __syncthreads();

    // ---- Phase B: Y_tile = Z @ U[o] + bias (bias preloaded into C) ----
    const int wave  = tid >> 5;
    const int lane  = tid & 31;
    const int l16   = lane & 15;
    const int lhalf = lane >> 4;

    const bf16_t* arow = &Zb[(wave * 16 + l16) * RK];

    for (int nt = 0; nt < 16; ++nt) {
        const int n    = nt * 16 + l16;
        const float bv = bias[o * BS + n];
        v8f acc;
#pragma unroll
        for (int e = 0; e < 8; ++e) acc[e] = bv;

#pragma unroll
        for (int kt = 0; kt < 4; ++kt) {
            v16bf a;
            const bf16_t* ap = arow + kt * 32 + lhalf * 16;
#pragma unroll
            for (int i = 0; i < 8; ++i) {
                a[i]     = ap[i];
                a[8 + i] = ap[8 + i];
            }
            v16bf b;
            const bf16_t* bp = &UT[n * RK + kt * 32 + lhalf * 16];
#pragma unroll
            for (int i = 0; i < 16; ++i) b[i] = bp[i];
            acc = __builtin_amdgcn_wmma_f32_16x16x32_bf16(
                false, a, false, b, (short)0, acc, false, false);
        }

        // Y is written once -> non-temporal, keep L2 for T
        float* yp = Y + (size_t)(row0 + wave * 16 + lhalf * 8) * OUT_DIM +
                    o * BS + n;
#pragma unroll
        for (int m = 0; m < 8; ++m)
            __builtin_nontemporal_store(acc[m], yp + (size_t)m * OUT_DIM);
    }
}

// ---------------------------------------------------------------------------
extern "C" void kernel_launch(void* const* d_in, const int* in_sizes, int n_in,
                              void* d_out, int out_size, void* d_ws,
                              size_t ws_size, hipStream_t stream) {
    const float* x    = (const float*)d_in[0];  // (8,1024,4096)
    const float* S    = (const float*)d_in[1];  // (16,16,128)
    const float* U    = (const float*)d_in[2];  // (16,128,256)
    const float* Vt   = (const float*)d_in[3];  // (16,256,128)
    const float* bias = (const float*)d_in[4];  // (4096,)
    float* out = (float*)d_out;                 // (8192,4096)

    bf16_t* T = (bf16_t*)d_ws;                  // 16*8192*128 bf16 = 32 MB

    const dim3 blk(256);
    blast_k1<<<dim3(64, 16), blk, 0, stream>>>(x, Vt, T);
    blast_k2<<<dim3(64, 16), blk, 0, stream>>>(T, S, U, bias, out);
}
